// TransformerLayerShardV1_76166950028538
// MI455X (gfx1250) — matrix operand order repaired and checked by reading, not compiled, not run on hardware
//
#include <hip/hip_runtime.h>
#include <hip/hip_bf16.h>

// ---------------------------------------------------------------------------
// TransformerLayerShard for MI455X (gfx1250), bf16 WMMA path.
// B=2, S=2048, D_MODEL=4096, H=16, DH=256, D_ROTARY=64.
// Workspace requirement: 256 MiB
//   [0,32MB)        xn (bf16)                 4096 x 4096
//   [32,160MB)      qkv (bf16) then ff1(bf16) 4096 x 12288 / 4096 x 16384
//   [160,192MB)     attn_vec (bf16)           4096 x 4096
//   [192,256MB)     attn_out (f32)            4096 x 4096
// ---------------------------------------------------------------------------

typedef unsigned short u16;
typedef __attribute__((ext_vector_type(16))) __bf16 v16bf;
typedef __attribute__((ext_vector_type(8)))  float  v8f;

// CDNA5 async global->LDS copy path (ASYNCcnt-tracked), if toolchain exposes it.
#if defined(__has_builtin)
#if __has_builtin(__builtin_amdgcn_global_load_async_to_lds_b128) && \
    __has_builtin(__builtin_amdgcn_s_wait_asynccnt)
#define USE_ASYNC_LDS 1
#endif
#endif
#ifndef USE_ASYNC_LDS
#define USE_ASYNC_LDS 0
#endif

#if USE_ASYNC_LDS
// Builtin signature (from diagnostics): (global v4i*, shared v4i*, imm int, imm int)
typedef int v4i_b128 __attribute__((vector_size(16)));
typedef __attribute__((address_space(1))) v4i_b128* async_gptr;
typedef __attribute__((address_space(3))) v4i_b128* async_lptr;
__device__ inline void async_cp_b128(const void* gsrc, void* lds_dst) {
  __builtin_amdgcn_global_load_async_to_lds_b128(
      (async_gptr)const_cast<void*>(gsrc),
      (async_lptr)lds_dst,
      /*offset=*/0, /*cpol=*/0);
}
#endif

union BF16Frag { v16bf v; uint4 u[2]; };

__device__ inline v16bf ld_frag(const u16* p0, const u16* p1) {
  BF16Frag f;
  f.u[0] = *(const uint4*)p0;
  f.u[1] = *(const uint4*)p1;
  return f.v;
}

__device__ inline v8f wmma_bf16(v16bf a, v16bf b, v8f c) {
  // D = A(16x32 bf16) * B(32x16 bf16) + C(16x16 f32)
  return __builtin_amdgcn_wmma_f32_16x16x32_bf16(
      /*neg_a=*/false, a, /*neg_b=*/false, b,
      /*c_mod=*/(short)0, c, /*reuse_a=*/false, /*reuse_b=*/false);
}

__device__ inline u16 f2bf(float x) {               // round-to-nearest-even
  unsigned int u = __builtin_bit_cast(unsigned int, x);
  unsigned int r = u + 0x7FFFu + ((u >> 16) & 1u);
  return (u16)(r >> 16);
}
__device__ inline float bf2f(u16 h) {
  unsigned int u = ((unsigned int)h) << 16;
  return __builtin_bit_cast(float, u);
}

template <int MASK>
__device__ inline float swz_xor(float v) {          // wave32 xor shuffle (16-lane groups)
  int i = __builtin_amdgcn_ds_swizzle(__builtin_bit_cast(int, v), (MASK << 10) | 31);
  return __builtin_bit_cast(float, i);
}
__device__ inline float rmax16(float v) {
  v = fmaxf(v, swz_xor<1>(v)); v = fmaxf(v, swz_xor<2>(v));
  v = fmaxf(v, swz_xor<4>(v)); v = fmaxf(v, swz_xor<8>(v));
  return v;
}
__device__ inline float rsum16(float v) {
  v += swz_xor<1>(v); v += swz_xor<2>(v);
  v += swz_xor<4>(v); v += swz_xor<8>(v);
  return v;
}

__device__ inline float gelu_f(float x) {           // jax.nn.gelu (tanh approx)
  float x3 = x * x * x;
  return 0.5f * x * (1.0f + tanhf(0.7978845608028654f * (x + 0.044715f * x3)));
}

#define DMODEL 4096
#define NHEAD  16
#define DHEAD  256
#define SEQ    2048
#define BATCH  2
#define QKVROW (NHEAD * 3 * DHEAD)   // 12288, row stride of qkv buffer

// ---------------------------------------------------------------------------
// 1) LayerNorm: one block per row; fp32 in, bf16 out.
// ---------------------------------------------------------------------------
__global__ __launch_bounds__(256) void ln_kernel(
    const float* __restrict__ x, const float* __restrict__ sc,
    const float* __restrict__ off, u16* __restrict__ xn) {
  __shared__ float red[256];
  int row = blockIdx.x;
  const float* xr = x + (size_t)row * DMODEL;
  float s = 0.f, ss = 0.f;
  for (int i = threadIdx.x; i < DMODEL; i += 256) {
    float v = xr[i]; s += v; ss += v * v;
  }
  red[threadIdx.x] = s; __syncthreads();
  for (int st = 128; st > 0; st >>= 1) {
    if (threadIdx.x < st) red[threadIdx.x] += red[threadIdx.x + st];
    __syncthreads();
  }
  float mean = red[0] * (1.0f / DMODEL);
  __syncthreads();
  red[threadIdx.x] = ss; __syncthreads();
  for (int st = 128; st > 0; st >>= 1) {
    if (threadIdx.x < st) red[threadIdx.x] += red[threadIdx.x + st];
    __syncthreads();
  }
  float var = red[0] * (1.0f / DMODEL) - mean * mean;
  float inv = rsqrtf(var + 1e-5f);
  for (int i = threadIdx.x; i < DMODEL; i += 256)
    xn[(size_t)row * DMODEL + i] = f2bf((xr[i] - mean) * inv * sc[i] + off[i]);
}

// ---------------------------------------------------------------------------
// 2) Tiled bf16 WMMA GEMM: C[M,N] = A[M,K](bf16) * B[K,N](f32 -> bf16 on load)
//    BM=BN=128, BK=32; 8 waves, each computes 64x32 (4x2 WMMA tiles).
//    A tile is double-buffered via CDNA5 async global->LDS DMA; fp32 weight
//    tile is converted to bf16 through VGPRs while the DMA is in flight.
//    EPI: 0 = store bf16; 1 = store f32; 2 = bias+gelu -> bf16;
//         3 = bias + addin(f32) -> f32.
// ---------------------------------------------------------------------------
#define BSTR 40   // padded k-stride of transposed B tile (units of u16)

template <int EPI>
__global__ __launch_bounds__(256) void gemm_bf16_kernel(
    const u16* __restrict__ A, const float* __restrict__ Bw,
    u16* __restrict__ Cb, float* __restrict__ Cf,
    const float* __restrict__ bias, const float* __restrict__ addin,
    int M, int N, int K) {
  __shared__ u16 sA[2][128 * 32];     // [m][k] row-major, double-buffered
  __shared__ u16 sBt[128 * BSTR];     // [n][k] (B transposed), padded

  int t = threadIdx.x;
  int wave = t >> 5, lane = t & 31, lh = lane & 15, kh = lane >> 4;
  int m0 = blockIdx.y * 128, n0 = blockIdx.x * 128;
  int wm = wave >> 2, wn = wave & 3;   // 2 x 4 wave grid

  // Per-thread A staging coordinates (2 x b128 per tile).
  int ar0 = (t + 0)   >> 2, ac0 = (t + 0)   & 3;
  int ar1 = (t + 256) >> 2, ac1 = (t + 256) & 3;

  v8f acc[4][2];
#pragma unroll
  for (int a = 0; a < 4; ++a)
#pragma unroll
    for (int b = 0; b < 2; ++b) acc[a][b] = (v8f){0.f,0.f,0.f,0.f,0.f,0.f,0.f,0.f};

#if USE_ASYNC_LDS
  // Prologue: kick off async DMA of A tile for ks = 0 into buffer 0.
  async_cp_b128(A + (size_t)(m0 + ar0) * K + ac0 * 8, &sA[0][ar0 * 32 + ac0 * 8]);
  async_cp_b128(A + (size_t)(m0 + ar1) * K + ac1 * 8, &sA[0][ar1 * 32 + ac1 * 8]);
#endif

  int buf = 0;
  for (int ks = 0; ks < K; ks += 32) {
    __syncthreads();   // prev-iter LDS reads done before overwriting sBt / sA[buf^1]
    // Stage B tile: load f32, convert to bf16, store transposed.
#pragma unroll
    for (int j = 0; j < 4; ++j) {
      int v = t + j * 256;            // 0..1023
      int kr = v >> 5, c = v & 31;
      const float* src = Bw + (size_t)(ks + kr) * N + n0 + c * 4;
      float4 w = *(const float4*)src;
      if (ks + 32 < K)                // warm L2/WGP$ for the next weight tile
        __builtin_prefetch(src + (size_t)32 * N, 0, 1);
      int col = c * 4;
      sBt[(col + 0) * BSTR + kr] = f2bf(w.x);
      sBt[(col + 1) * BSTR + kr] = f2bf(w.y);
      sBt[(col + 2) * BSTR + kr] = f2bf(w.z);
      sBt[(col + 3) * BSTR + kr] = f2bf(w.w);
    }
#if USE_ASYNC_LDS
    if (ks + 32 < K) {
      // Issue DMA for the next A tile, then wait for the current one
      // (async loads complete in order: <=2 outstanding keeps only the new pair).
      const u16* An = A + (size_t)m0 * K + (ks + 32);
      async_cp_b128(An + (size_t)ar0 * K + ac0 * 8, &sA[buf ^ 1][ar0 * 32 + ac0 * 8]);
      async_cp_b128(An + (size_t)ar1 * K + ac1 * 8, &sA[buf ^ 1][ar1 * 32 + ac1 * 8]);
      __builtin_amdgcn_s_wait_asynccnt(2);
    } else {
      __builtin_amdgcn_s_wait_asynccnt(0);
    }
#else
    {
      const u16* As = A + (size_t)m0 * K + ks;
      *(uint4*)&sA[buf][ar0 * 32 + ac0 * 8] = *(const uint4*)(As + (size_t)ar0 * K + ac0 * 8);
      *(uint4*)&sA[buf][ar1 * 32 + ac1 * 8] = *(const uint4*)(As + (size_t)ar1 * K + ac1 * 8);
    }
#endif
    __syncthreads();

    v16bf af[4], bfr[2];
#pragma unroll
    for (int mf = 0; mf < 4; ++mf) {
      const u16* p = &sA[buf][(wm * 64 + mf * 16 + lh) * 32 + kh * 8];
      af[mf] = ld_frag(p, p + 16);    // K chunks {kh*8..}, {16+kh*8..}
    }
#pragma unroll
    for (int nf = 0; nf < 2; ++nf) {
      const u16* p = &sBt[(wn * 32 + nf * 16 + lh) * BSTR + kh * 8];
      bfr[nf] = ld_frag(p, p + 16);   // K chunks {kh*8..}, {16+kh*8..}
    }
#pragma unroll
    for (int mf = 0; mf < 4; ++mf)
#pragma unroll
      for (int nf = 0; nf < 2; ++nf)
        acc[mf][nf] = wmma_bf16(af[mf], bfr[nf], acc[mf][nf]);
    buf ^= 1;
  }

  // Epilogue (C layout: VGPR i -> row i / i+8 by lane-half; col = lane&15)
#pragma unroll
  for (int mf = 0; mf < 4; ++mf) {
#pragma unroll
    for (int nf = 0; nf < 2; ++nf) {
      int col = n0 + wn * 32 + nf * 16 + lh;
      float bval = (EPI >= 2) ? bias[col] : 0.f;
#pragma unroll
      for (int i = 0; i < 8; ++i) {
        int row = m0 + wm * 64 + mf * 16 + i + kh * 8;
        float v = acc[mf][nf][i];
        if (EPI == 0) {
          Cb[(size_t)row * N + col] = f2bf(v);
        } else if (EPI == 1) {
          Cf[(size_t)row * N + col] = v;
        } else if (EPI == 2) {
          Cb[(size_t)row * N + col] = f2bf(gelu_f(v + bval));
        } else {
          Cf[(size_t)row * N + col] = v + bval + addin[(size_t)row * N + col];
        }
      }
    }
  }
}

// ---------------------------------------------------------------------------
// 3) RoPE, in-place on the bf16 qkv buffer [B,S,H, 3*DHEAD] (q|v|k).
// ---------------------------------------------------------------------------
__global__ __launch_bounds__(256) void rope_kernel(u16* __restrict__ qkv) {
  int idx = blockIdx.x * 256 + threadIdx.x;   // B*S*H*32 threads, one rot pair each
  int j = idx & 31;
  int h = (idx >> 5) & (NHEAD - 1);
  int s = (idx >> 9) & (SEQ - 1);
  int b = idx >> 20;
  size_t base = (((size_t)(b * SEQ + s)) * NHEAD + h) * (3 * DHEAD);
  float invf = __expf(-(float)j * (9.210340371976184f / 32.f));  // 10000^(-j/32)
  float ang = (float)s * invf;
  float sn, cs;
  __sincosf(ang, &sn, &cs);
  // q at offset 0
  {
    float x1 = bf2f(qkv[base + 2 * j]), x2 = bf2f(qkv[base + 2 * j + 1]);
    qkv[base + 2 * j]     = f2bf(x1 * cs - x2 * sn);
    qkv[base + 2 * j + 1] = f2bf(x2 * cs + x1 * sn);
  }
  // k at offset 2*DHEAD = 512
  {
    size_t kb = base + 2 * DHEAD;
    float x1 = bf2f(qkv[kb + 2 * j]), x2 = bf2f(qkv[kb + 2 * j + 1]);
    qkv[kb + 2 * j]     = f2bf(x1 * cs - x2 * sn);
    qkv[kb + 2 * j + 1] = f2bf(x2 * cs + x1 * sn);
  }
}

// ---------------------------------------------------------------------------
// 4) Flash attention: block = (b,h) x 128-query tile; 8 waves x 16 rows.
//    Q fragments in registers; K tile DMA'd straight to LDS (async path),
//    V tile staged transposed; both matmuls via bf16 WMMA; streaming softmax.
// ---------------------------------------------------------------------------
#define VSTR 40   // padded key-stride of transposed V tile

__global__ __launch_bounds__(256) void attn_kernel(
    const u16* __restrict__ qkv, const float* __restrict__ attn_bias,
    u16* __restrict__ attn_vec) {
  __shared__ u16 sK[32 * DHEAD];        // [key][d]   (k rows straight copy)
  __shared__ u16 sVt[DHEAD * VSTR];     // [d][key]   (v transposed)
  __shared__ u16 sP[8 * 16 * 32];       // per-wave P relayout scratch

  int bh = blockIdx.x;
  int b = bh >> 4, h = bh & (NHEAD - 1);
  int q0 = blockIdx.y * 128;
  int t = threadIdx.x, wave = t >> 5, lane = t & 31, lh = lane & 15, kh = lane >> 4;
  const int rowstride = QKVROW;                                  // 12288
  size_t head_base = (((size_t)b * SEQ) * NHEAD + h) * (3 * DHEAD);

  // Q fragments for this wave's 16 rows (A-matrix layout), kept in registers.
  int qw0 = q0 + wave * 16;
  const u16* qptr = qkv + head_base + (size_t)(qw0 + lh) * rowstride;
  v16bf qf[8];
#pragma unroll
  for (int dc = 0; dc < 8; ++dc) {
    const u16* p = qptr + dc * 32 + kh * 16;
    qf[dc] = ld_frag(p, p + 8);
  }

  v8f acc[16];
#pragma unroll
  for (int nf = 0; nf < 16; ++nf) acc[nf] = (v8f){0.f,0.f,0.f,0.f,0.f,0.f,0.f,0.f};
  float m[8], l[8];
#pragma unroll
  for (int i = 0; i < 8; ++i) { m[i] = -3.0e38f; l[i] = 0.f; }

  for (int kb = 0; kb < q0 + 128; kb += 32) {
    __syncthreads();
    // Stage K (offset 2*DHEAD) straight (async DMA when available);
    // stage V (offset DHEAD) transposed through VGPRs.
#pragma unroll
    for (int i = 0; i < 4; ++i) {
      int v = t + i * 256;               // 0..1023
      int key = v >> 5, c = v & 31;      // c: uint4 (8 bf16) chunk along d
      const u16* src = qkv + head_base + (size_t)(kb + key) * rowstride;
#if USE_ASYNC_LDS
      async_cp_b128(src + 2 * DHEAD + c * 8, &sK[key * DHEAD + c * 8]);
#else
      *(uint4*)&sK[key * DHEAD + c * 8] = *(const uint4*)(src + 2 * DHEAD + c * 8);
#endif
      uint4 vv = *(const uint4*)(src + DHEAD + c * 8);
      const u16* pv = (const u16*)&vv;
      int d0 = c * 8;
#pragma unroll
      for (int q = 0; q < 8; ++q) sVt[(d0 + q) * VSTR + key] = pv[q];
    }
#if USE_ASYNC_LDS
    __builtin_amdgcn_s_wait_asynccnt(0);
#endif
    __syncthreads();

    bool active = (kb <= qw0 + 15);      // wave-uniform causal skip
    if (active) {
      // S = Q * K^T for two 16-key chunks (K-dim = DHEAD, 8 WMMA steps each),
      // with fragment loads software-pipelined one step ahead.
      v8f s0 = (v8f){0.f,0.f,0.f,0.f,0.f,0.f,0.f,0.f};
      v8f s1 = s0;
      const u16* p0 = &sK[(0  + lh) * DHEAD + kh * 16];
      const u16* p1 = &sK[(16 + lh) * DHEAD + kh * 16];
      v16bf k0 = ld_frag(p0, p0 + 8);
      v16bf k1 = ld_frag(p1, p1 + 8);
#pragma unroll
      for (int dc = 0; dc < 8; ++dc) {
        v16bf nk0 = k0, nk1 = k1;
        if (dc < 7) {
          const u16* a0 = &sK[(0  + lh) * DHEAD + (dc + 1) * 32 + kh * 16];
          const u16* a1 = &sK[(16 + lh) * DHEAD + (dc + 1) * 32 + kh * 16];
          nk0 = ld_frag(a0, a0 + 8);
          nk1 = ld_frag(a1, a1 + 8);
        }
        s0 = wmma_bf16(qf[dc], k0, s0);
        s1 = wmma_bf16(qf[dc], k1, s1);
        k0 = nk0; k1 = nk1;
      }
      // Streaming softmax over the 32 staged keys.
      u16* sp = &sP[wave * 512];
#pragma unroll
      for (int i = 0; i < 8; ++i) {
        int q   = qw0 + i + kh * 8;
        int k0i = kb + lh, k1i = kb + 16 + lh;
        float t0 = s0[i] * 0.0625f + attn_bias[(size_t)q * SEQ + k0i] +
                   (k0i > q ? -1.0e10f : 0.f);
        float t1 = s1[i] * 0.0625f + attn_bias[(size_t)q * SEQ + k1i] +
                   (k1i > q ? -1.0e10f : 0.f);
        float nm = fmaxf(m[i], rmax16(fmaxf(t0, t1)));
        float scl = __expf(m[i] - nm);
        float p0v = __expf(t0 - nm), p1v = __expf(t1 - nm);
        l[i] = l[i] * scl + rsum16(p0v + p1v);
        m[i] = nm;
        int prow = i + kh * 8;
        sp[prow * 32 + lh]      = f2bf(p0v);
        sp[prow * 32 + 16 + lh] = f2bf(p1v);
#pragma unroll
        for (int nf = 0; nf < 16; ++nf) acc[nf][i] *= scl;
      }
      // acc += P(16x32) * V(32x256)
      const u16* pp = &sp[lh * 32 + kh * 16];
      v16bf pf = ld_frag(pp, pp + 8);
#pragma unroll
      for (int nf = 0; nf < 16; ++nf) {
        const u16* pv = &sVt[(nf * 16 + lh) * VSTR + kh * 16];
        acc[nf] = wmma_bf16(pf, ld_frag(pv, pv + 8), acc[nf]);
      }
    }
  }

  // Normalize and store attn_vec[b, s, h*DHEAD + d] as bf16.
#pragma unroll
  for (int i = 0; i < 8; ++i) {
    int q = qw0 + i + kh * 8;
    float inv = 1.0f / l[i];
    size_t rowb = ((size_t)(b * SEQ + q)) * DMODEL + h * DHEAD;
#pragma unroll
    for (int nf = 0; nf < 16; ++nf)
      attn_vec[rowb + nf * 16 + lh] = f2bf(acc[nf][i] * inv);
  }
}

// ---------------------------------------------------------------------------
// Host-side launcher
// ---------------------------------------------------------------------------
extern "C" void kernel_launch(void* const* d_in, const int* in_sizes, int n_in,
                              void* d_out, int out_size, void* d_ws, size_t ws_size,
                              hipStream_t stream) {
  const float* x         = (const float*)d_in[0];
  const float* attn_bias = (const float*)d_in[1];
  const float* ln_scale  = (const float*)d_in[2];
  const float* ln_offset = (const float*)d_in[3];
  const float* w_qkv     = (const float*)d_in[4];
  const float* w_o       = (const float*)d_in[5];
  const float* w_in      = (const float*)d_in[6];
  const float* b_in      = (const float*)d_in[7];
  const float* w_out     = (const float*)d_in[8];
  const float* b_out     = (const float*)d_in[9];
  float* out = (float*)d_out;

  char* ws = (char*)d_ws;
  u16*   xn       = (u16*)(ws);                                  // 32 MB
  u16*   big      = (u16*)(ws + (size_t)32 * 1024 * 1024);       // 128 MB (qkv -> ff1)
  u16*   attn_vec = (u16*)(ws + (size_t)160 * 1024 * 1024);      // 32 MB
  float* attn_out = (float*)(ws + (size_t)192 * 1024 * 1024);    // 64 MB

  // 1) LayerNorm
  ln_kernel<<<BATCH * SEQ, 256, 0, stream>>>(x, ln_scale, ln_offset, xn);
  // 2) qkv = xn @ w_qkv  (M=4096, N=12288, K=4096), bf16 out
  gemm_bf16_kernel<0><<<dim3(QKVROW / 128, (BATCH * SEQ) / 128), 256, 0, stream>>>(
      xn, w_qkv, big, nullptr, nullptr, nullptr, BATCH * SEQ, QKVROW, DMODEL);
  // 3) RoPE in place on q and k
  rope_kernel<<<(BATCH * SEQ * NHEAD * 32) / 256, 256, 0, stream>>>(big);
  // 4) attention -> attn_vec (bf16)
  attn_kernel<<<dim3(BATCH * NHEAD, SEQ / 128), 256, 0, stream>>>(big, attn_bias, attn_vec);
  // 5) attn_out = attn_vec @ w_o (f32 out)
  gemm_bf16_kernel<1><<<dim3(DMODEL / 128, (BATCH * SEQ) / 128), 256, 0, stream>>>(
      attn_vec, w_o, nullptr, attn_out, nullptr, nullptr, BATCH * SEQ, DMODEL, DMODEL);
  // 6) ff1 = gelu(xn @ w_in + b_in) (bf16, reuses qkv region)
  gemm_bf16_kernel<2><<<dim3((4 * DMODEL) / 128, (BATCH * SEQ) / 128), 256, 0, stream>>>(
      xn, w_in, big, nullptr, b_in, nullptr, BATCH * SEQ, 4 * DMODEL, DMODEL);
  // 7) out = ff1 @ w_out + b_out + attn_out (f32)
  gemm_bf16_kernel<3><<<dim3(DMODEL / 128, (BATCH * SEQ) / 128), 256, 0, stream>>>(
      big, w_out, nullptr, out, b_out, attn_out, BATCH * SEQ, DMODEL, 4 * DMODEL);
}
